// MultiHeadAttention_90391881712132
// MI455X (gfx1250) — matrix operand order repaired and checked by reading, not compiled, not run on hardware
//
#include <hip/hip_runtime.h>
#include <hip/hip_bf16.h>

#define DEV __device__ __forceinline__

typedef __attribute__((ext_vector_type(16))) __bf16       v16bf;
typedef __attribute__((ext_vector_type(8)))  float        v8f;
typedef __attribute__((ext_vector_type(4)))  float        f32x4;
typedef __attribute__((ext_vector_type(4)))  unsigned int u32x4;
typedef unsigned short u16;

union Frag16 { u32x4 q[2]; v16bf v; };

constexpr int Bb = 4, Ls = 1024, Dm = 1024, Hh = 16, DHd = 64;

DEV u16 f2bf(float f) {
  union { float f; unsigned int u; } v; v.f = f;
  unsigned int r = v.u + 0x7FFFu + ((v.u >> 16) & 1u);
  return (u16)(r >> 16);
}

DEV v8f wmma_bf16(const Frag16& a, const Frag16& b, v8f c) {
  // D = A(16x32 bf16) * B(32x16 bf16) + C(f32)
  return __builtin_amdgcn_wmma_f32_16x16x32_bf16(false, a.v, false, b.v,
                                                 (short)0, c, false, false);
}

// Async 16B copy global -> LDS (CDNA5, tracked by ASYNCcnt).
DEV void async_cp16(const u16* g, const u16* l) {
  unsigned lds = (unsigned)(unsigned long long)l;  // LDS aperture: addr[31:0]
  asm volatile("global_load_async_to_lds_b128 %0, %1, off"
               :: "v"(lds), "v"(g) : "memory");
}
DEV void wait_async0() { asm volatile("s_wait_asynccnt 0" ::: "memory"); }

// ---------------------------------------------------------------------------
// Kernel 0a: bulk f32 -> bf16 row-major convert (Q, K, V inputs)
// ---------------------------------------------------------------------------
__global__ __launch_bounds__(256) void k_cvt(const float* __restrict__ src,
                                             u16* __restrict__ dst) {
  size_t i = ((size_t)blockIdx.x * 256 + threadIdx.x) * 8;
  f32x4 a = *(const f32x4*)(src + i);
  f32x4 b = *(const f32x4*)(src + i + 4);
  u32x4 o;
  o.x = (unsigned)f2bf(a.x) | ((unsigned)f2bf(a.y) << 16);
  o.y = (unsigned)f2bf(a.z) | ((unsigned)f2bf(a.w) << 16);
  o.z = (unsigned)f2bf(b.x) | ((unsigned)f2bf(b.y) << 16);
  o.w = (unsigned)f2bf(b.z) | ((unsigned)f2bf(b.w) << 16);
  *(u32x4*)(dst + i) = o;
}

// ---------------------------------------------------------------------------
// Kernel 0b: K_mask int -> additive f32 bias (0 or -1e9)
// ---------------------------------------------------------------------------
__global__ __launch_bounds__(256) void k_mask(const int* __restrict__ km,
                                              float* __restrict__ mb) {
  int i = blockIdx.x * 256 + threadIdx.x;
  mb[i] = km[i] ? 0.f : -1e9f;
}

// ---------------------------------------------------------------------------
// Kernel 1: weight f32 [K][N] -> bf16 transposed [N][K]
// ---------------------------------------------------------------------------
__global__ void k_wt(const float* __restrict__ W, u16* __restrict__ Wt) {
  __shared__ float t[32][33];
  int n0 = blockIdx.x * 32, k0 = blockIdx.y * 32;
  int tx = threadIdx.x, ty = threadIdx.y;
  for (int i = ty; i < 32; i += 8)
    t[i][tx] = W[(size_t)(k0 + i) * Dm + n0 + tx];
  __syncthreads();
  for (int i = ty; i < 32; i += 8)
    Wt[(size_t)(n0 + i) * Dm + k0 + tx] = f2bf(t[tx][i]);
}

// ---------------------------------------------------------------------------
// Kernel 2: WMMA GEMM  C[4096x1024] = Ab(bf16) * Wt^T + bias, then *oscale
//   MODE 0: out = bf16 head-major [B,H,L,DH]            (q, k)
//   MODE 1: out = bf16 head-major transposed [B,H,DH,L] (v)
//   MODE 2: out = f32 row-major                         (out-proj)
// 256 thr / 8 waves in 4x2 grid; wave -> 32 rows x 64 cols.
// Tile BM=128 BN=128 BK=64; double-buffered LDS, async global->LDS fills.
// ---------------------------------------------------------------------------
template <int MODE>
__global__ __launch_bounds__(256) void k_gemm(const u16* __restrict__ Ab,
                                              const u16* __restrict__ Bt,
                                              const float* __restrict__ bias,
                                              void* __restrict__ outp,
                                              float oscale) {
  __shared__ __align__(16) u16 lA[2][128 * 64];  // [row][k] bf16, 16KB/buf
  __shared__ __align__(16) u16 lB[2][128 * 64];  // [n][k]  bf16
  const int tid = threadIdx.x;
  const int m0 = blockIdx.y * 128, n0 = blockIdx.x * 128;
  const int w = tid >> 5, lane = tid & 31, half = lane >> 4, ln = lane & 15;
  const int rw = w >> 1, cw = w & 1;  // wave tile: rows rw*32.., cols cw*64..

  // Tile = 128x64 bf16 = 1024 x 16B chunks; 4 chunks/thread per matrix.
  auto issue = [&](int buf, int k0) {
#pragma unroll
    for (int p = 0; p < 4; ++p) {
      int id = tid * 4 + p;
      int r = id >> 3, kq = (id & 7) * 8;
      async_cp16(Ab + (size_t)(m0 + r) * Dm + k0 + kq, &lA[buf][r * 64 + kq]);
      async_cp16(Bt + (size_t)(n0 + r) * Dm + k0 + kq, &lB[buf][r * 64 + kq]);
    }
  };

  v8f zero = {0.f, 0.f, 0.f, 0.f, 0.f, 0.f, 0.f, 0.f};
  v8f acc[8];  // [s*4 + c]: s = row sub-strip (16), c = col group (16)
#pragma unroll
  for (int i = 0; i < 8; ++i) acc[i] = zero;

  constexpr int NK = Dm / 64;  // 16 staging rounds
  issue(0, 0);
  for (int kt = 0; kt < NK; ++kt) {
    const int cur = kt & 1;
    wait_async0();      // my async writes for this round landed in LDS
    __syncthreads();    // everyone's have
    if (kt + 1 < NK) issue(1 - cur, (kt + 1) * 64);  // prefetch overlaps math

#pragma unroll
    for (int kc = 0; kc < 2; ++kc) {  // two K=32 chunks per round
      // A frags: lane<16 row=ln K=0..15; lane>=16 K=16..31 (contiguous)
      Frag16 a[2];
#pragma unroll
      for (int s = 0; s < 2; ++s) {
        int ar = rw * 32 + s * 16 + ln;
        a[s].q[0] = *(const u32x4*)&lA[cur][ar * 64 + kc * 32 + half * 16];
        a[s].q[1] = *(const u32x4*)&lA[cur][ar * 64 + kc * 32 + half * 16 + 8];
      }
      // B frags: lane n -> col n, K = half*16 .. +15 (contiguous)
      Frag16 bf[4];
#pragma unroll
      for (int c = 0; c < 4; ++c) {
        int bn = cw * 64 + c * 16 + ln;
        bf[c].q[0] = *(const u32x4*)&lB[cur][bn * 64 + kc * 32 + half * 16];
        bf[c].q[1] = *(const u32x4*)&lB[cur][bn * 64 + kc * 32 + half * 16 + 8];
      }
#pragma unroll
      for (int s = 0; s < 2; ++s)
#pragma unroll
        for (int c = 0; c < 4; ++c)
          acc[s * 4 + c] = wmma_bf16(a[s], bf[c], acc[s * 4 + c]);
    }
    __syncthreads();    // done reading buf[cur]; safe to refill next round
  }

  // Epilogue: C slot j -> row +j+8*half, col +ln
#pragma unroll
  for (int s = 0; s < 2; ++s) {
#pragma unroll
    for (int c = 0; c < 4; ++c) {
      int col = n0 + cw * 64 + c * 16 + ln;
      float bval = bias[col];
#pragma unroll
      for (int j = 0; j < 8; ++j) {
        int m = m0 + rw * 32 + s * 16 + j + 8 * half;
        float val = (acc[s * 4 + c][j] + bval) * oscale;
        if constexpr (MODE == 2) {
          ((float*)outp)[(size_t)m * Dm + col] = val;
        } else {
          int b = m >> 10, l = m & 1023;
          int h = col >> 6, dcol = col & 63;
          u16* o = (u16*)outp;
          if constexpr (MODE == 0)
            o[(((size_t)(b * Hh + h)) * Ls + l) * DHd + dcol] = f2bf(val);
          else
            o[(((size_t)(b * Hh + h)) * DHd + dcol) * Ls + l] = f2bf(val);
        }
      }
    }
  }
}

// ---------------------------------------------------------------------------
// Kernel 3: flash attention per (b,h). Block=128 thr (4 waves), each wave a
// 16-row q strip; online softmax over key tiles of 64. q pre-scaled by 1/8.
// q,k: bf16 [B,H,L,64]; v: bf16 [B,H,64,L]; mask: additive f32 [B,L]
// ctx out: bf16 [B,L,D]
// ---------------------------------------------------------------------------
__global__ __launch_bounds__(128) void k_attn(const u16* __restrict__ qh,
                                              const u16* __restrict__ kh,
                                              const u16* __restrict__ vt,
                                              const float* __restrict__ mbias,
                                              u16* __restrict__ ctx) {
  __shared__ __align__(16) u16 pL[4 * 16 * 64];  // per-wave P staging
  const int tid = threadIdx.x;
  const int w = tid >> 5, lane = tid & 31, half = lane >> 4, ln = lane & 15;
  const int bh = blockIdx.y, b = bh >> 4, h = bh & 15;
  const int q0 = blockIdx.x * 64 + w * 16;

  // q A-fragments for K(dh) = 0..31 and 32..63
  Frag16 aq[2];
  const u16* qrow = qh + ((size_t)bh * Ls + q0 + ln) * DHd;
#pragma unroll
  for (int kk = 0; kk < 2; ++kk) {
    aq[kk].q[0] = *(const u32x4*)(qrow + kk * 32 + half * 16);
    aq[kk].q[1] = *(const u32x4*)(qrow + kk * 32 + half * 16 + 8);
  }

  v8f zero = {0.f, 0.f, 0.f, 0.f, 0.f, 0.f, 0.f, 0.f};
  v8f ctxa[4];
#pragma unroll
  for (int c = 0; c < 4; ++c) ctxa[c] = zero;
  float mrow[8], srow[8];
#pragma unroll
  for (int j = 0; j < 8; ++j) { mrow[j] = -3e38f; srow[j] = 0.f; }

  u16* pw = pL + w * (16 * 64);

  for (int kt = 0; kt < Ls / 64; ++kt) {
    // ---- scores: S[16q x 64keys] as 4 fragments ----
    v8f sc[4];
#pragma unroll
    for (int c = 0; c < 4; ++c) {
      int key = kt * 64 + c * 16 + ln;
      const u16* krow = kh + ((size_t)bh * Ls + key) * DHd;
      Frag16 b0, b1;  // B = k^T: lane -> key col, K(dh) contiguous
      b0.q[0] = *(const u32x4*)(krow + half * 16);
      b0.q[1] = *(const u32x4*)(krow + half * 16 + 8);
      b1.q[0] = *(const u32x4*)(krow + 32 + half * 16);
      b1.q[1] = *(const u32x4*)(krow + 32 + half * 16 + 8);
      v8f s = wmma_bf16(aq[0], b0, zero);
      s = wmma_bf16(aq[1], b1, s);
      float mb = mbias[b * Ls + key];  // 0 or -1e9, additive mask
#pragma unroll
      for (int j = 0; j < 8; ++j)
        sc[c][j] = s[j] + mb;
    }
    // ---- online softmax: row stats across lanes 0..15 / 16..31 ----
#pragma unroll
    for (int j = 0; j < 8; ++j) {
      float rm = fmaxf(fmaxf(sc[0][j], sc[1][j]), fmaxf(sc[2][j], sc[3][j]));
      rm = fmaxf(rm, __shfl_xor(rm, 1));
      rm = fmaxf(rm, __shfl_xor(rm, 2));
      rm = fmaxf(rm, __shfl_xor(rm, 4));
      rm = fmaxf(rm, __shfl_xor(rm, 8));
      float mn = fmaxf(mrow[j], rm);
      float al = __expf(mrow[j] - mn);
      mrow[j] = mn;
      float p0 = __expf(sc[0][j] - mn), p1 = __expf(sc[1][j] - mn);
      float p2 = __expf(sc[2][j] - mn), p3 = __expf(sc[3][j] - mn);
      sc[0][j] = p0; sc[1][j] = p1; sc[2][j] = p2; sc[3][j] = p3;
      float rs = p0 + p1 + p2 + p3;
      rs += __shfl_xor(rs, 1);
      rs += __shfl_xor(rs, 2);
      rs += __shfl_xor(rs, 4);
      rs += __shfl_xor(rs, 8);
      srow[j] = srow[j] * al + rs;
      ctxa[0][j] *= al; ctxa[1][j] *= al; ctxa[2][j] *= al; ctxa[3][j] *= al;
    }
    // ---- reshape P (C-layout) -> A-layout via wave-private LDS ----
#pragma unroll
    for (int c = 0; c < 4; ++c)
#pragma unroll
      for (int j = 0; j < 8; ++j)
        pw[(j + 8 * half) * 64 + c * 16 + ln] = f2bf(sc[c][j]);
    __asm__ volatile("s_wait_dscnt 0" ::: "memory");
    Frag16 ap[2];
#pragma unroll
    for (int kk = 0; kk < 2; ++kk) {
      ap[kk].q[0] = *(const u32x4*)&pw[ln * 64 + kk * 32 + half * 16];
      ap[kk].q[1] = *(const u32x4*)&pw[ln * 64 + kk * 32 + half * 16 + 8];
    }
    // ---- ctx += P * V  (B from v^T: lane -> dh col, keys contiguous) ----
#pragma unroll
    for (int c = 0; c < 4; ++c) {
#pragma unroll
      for (int kk = 0; kk < 2; ++kk) {
        const u16* vrow = vt + ((size_t)bh * DHd + c * 16 + ln) * Ls +
                          kt * 64 + kk * 32 + half * 16;
        Frag16 bv;
        bv.q[0] = *(const u32x4*)(vrow);
        bv.q[1] = *(const u32x4*)(vrow + 8);
        ctxa[c] = wmma_bf16(ap[kk], bv, ctxa[c]);
      }
    }
  }
  // ---- normalize + store ctx bf16 [B,L,D] ----
#pragma unroll
  for (int c = 0; c < 4; ++c) {
    int col = h * DHd + c * 16 + ln;
#pragma unroll
    for (int j = 0; j < 8; ++j) {
      int row = q0 + j + 8 * half;
      float val = ctxa[c][j] / srow[j];
      ctx[((size_t)b * Ls + row) * Dm + col] = f2bf(val);
    }
  }
}

// ---------------------------------------------------------------------------
// Kernel 4: residual + layernorm. One block per row, 4 elems/thread.
// ---------------------------------------------------------------------------
__global__ __launch_bounds__(256) void k_ln(const float* __restrict__ ao,
                                            const float* __restrict__ Qin,
                                            const float* __restrict__ g,
                                            const float* __restrict__ bb,
                                            float* __restrict__ out) {
  __shared__ float r1[256], r2[256];
  const int row = blockIdx.x, t = threadIdx.x;
  f32x4 a = *(const f32x4*)(ao + (size_t)row * Dm + t * 4);
  f32x4 q = *(const f32x4*)(Qin + (size_t)row * Dm + t * 4);
  f32x4 x = a + q;
  r1[t] = x.x + x.y + x.z + x.w;
  r2[t] = x.x * x.x + x.y * x.y + x.z * x.z + x.w * x.w;
  __syncthreads();
  for (int o = 128; o > 0; o >>= 1) {
    if (t < o) { r1[t] += r1[t + o]; r2[t] += r2[t + o]; }
    __syncthreads();
  }
  float mu = r1[0] * (1.0f / Dm);
  float var = r2[0] * (1.0f / Dm) - mu * mu;
  float inv = rsqrtf(var + 1e-5f);
  f32x4 gv = *(const f32x4*)(g + t * 4);
  f32x4 bv = *(const f32x4*)(bb + t * 4);
  f32x4 o4;
  o4.x = (x.x - mu) * inv * gv.x + bv.x;
  o4.y = (x.y - mu) * inv * gv.y + bv.y;
  o4.z = (x.z - mu) * inv * gv.z + bv.z;
  o4.w = (x.w - mu) * inv * gv.w + bv.w;
  *(f32x4*)(out + (size_t)row * Dm + t * 4) = o4;
}

// ---------------------------------------------------------------------------
extern "C" void kernel_launch(void* const* d_in, const int* in_sizes, int n_in,
                              void* d_out, int out_size, void* d_ws,
                              size_t ws_size, hipStream_t stream) {
  (void)in_sizes; (void)n_in; (void)out_size; (void)ws_size;
  const float* Q    = (const float*)d_in[0];
  const float* Kin  = (const float*)d_in[1];
  const float* V    = (const float*)d_in[2];
  const int*   kmsk = (const int*)d_in[3];
  const float* WQ   = (const float*)d_in[4];
  const float* bQ   = (const float*)d_in[5];
  const float* WK   = (const float*)d_in[6];
  const float* bK   = (const float*)d_in[7];
  const float* WV   = (const float*)d_in[8];
  const float* bV   = (const float*)d_in[9];
  const float* WO   = (const float*)d_in[10];
  const float* bO   = (const float*)d_in[11];
  const float* lg   = (const float*)d_in[12];
  const float* lb   = (const float*)d_in[13];

  char* ws = (char*)d_ws;
  const size_t MB = 1ull << 20;
  u16*   WQt = (u16*)(ws + 0 * MB);    // 2MB each (1024*1024 bf16)
  u16*   WKt = (u16*)(ws + 2 * MB);
  u16*   WVt = (u16*)(ws + 4 * MB);
  u16*   WOt = (u16*)(ws + 6 * MB);
  u16*   Qb  = (u16*)(ws + 8 * MB);    // bf16 row-major inputs, 8MB each
  u16*   Kb  = (u16*)(ws + 16 * MB);
  u16*   Vb  = (u16*)(ws + 24 * MB);
  u16*   qh  = (u16*)(ws + 32 * MB);   // [B,H,L,DH]
  u16*   kh  = (u16*)(ws + 40 * MB);
  u16*   vt  = (u16*)(ws + 48 * MB);   // [B,H,DH,L]
  float* mb  = (float*)(ws + 56 * MB); // additive mask bias [B,L] (16KB)
  u16*   ctx = (u16*)(ws + 8 * MB);    // reuse Qb (dead after projections)
  float* ao  = (float*)(ws + 16 * MB); // reuse Kb+Vb (16MB)

  const int Ntot = Bb * Ls * Dm;       // 4M elements
  k_cvt<<<Ntot / (256 * 8), 256, 0, stream>>>(Q, Qb);
  k_cvt<<<Ntot / (256 * 8), 256, 0, stream>>>(Kin, Kb);
  k_cvt<<<Ntot / (256 * 8), 256, 0, stream>>>(V, Vb);
  k_mask<<<(Bb * Ls) / 256, 256, 0, stream>>>(kmsk, mb);

  dim3 bt(32, 8), gt(32, 32);
  k_wt<<<gt, bt, 0, stream>>>(WQ, WQt);
  k_wt<<<gt, bt, 0, stream>>>(WK, WKt);
  k_wt<<<gt, bt, 0, stream>>>(WV, WVt);
  k_wt<<<gt, bt, 0, stream>>>(WO, WOt);

  dim3 gg(Dm / 128, (Bb * Ls) / 128);  // (8, 32)
  // q carries the 1/sqrt(DH)=0.125 softmax scale (exact power of two)
  k_gemm<0><<<gg, 256, 0, stream>>>(Qb, WQt, bQ, qh, 0.125f);
  k_gemm<0><<<gg, 256, 0, stream>>>(Kb, WKt, bK, kh, 1.0f);
  k_gemm<1><<<gg, 256, 0, stream>>>(Vb, WVt, bV, vt, 1.0f);

  dim3 ga(Ls / 64, Bb * Hh);           // (16, 64)
  k_attn<<<ga, 128, 0, stream>>>(qh, kh, vt, mb, ctx);

  k_gemm<2><<<gg, 256, 0, stream>>>(ctx, WOt, bO, ao, 1.0f);

  k_ln<<<Bb * Ls, 256, 0, stream>>>(ao, Q, lg, lb, (float*)d_out);
}
